// NattenBlock_9053791060626
// MI455X (gfx1250) — compile-verified, matrix operand order checked by reading, not run on hardware
//
#include <hip/hip_runtime.h>
#include <math.h>

// ---------------------------------------------------------------------------
// Types for CDNA5 WMMA (gfx1250, wave32)
// ---------------------------------------------------------------------------
typedef __bf16 bf16_t;
typedef __attribute__((ext_vector_type(16))) __bf16 v16bf;
typedef __attribute__((ext_vector_type(8)))  __bf16 v8bf;
typedef __attribute__((ext_vector_type(8)))  float  v8f;

#define D_MODEL 512
#define D_HEAD  64
#define N_HEADS 8
#define KSIZE   7
#define NB      (KSIZE * KSIZE)
#define HIMG    48
#define WIMG    48
#define NBATCH  2
#define NPIX    (NBATCH * HIMG * WIMG)   // 4608

// ---------------------------------------------------------------------------
// fp32 -> bf16 conversion kernels (RNE via clang __bf16 cast)
// ---------------------------------------------------------------------------
__global__ void natten_f32_to_bf16(const float* __restrict__ in,
                                   bf16_t* __restrict__ out, int n) {
    int i = blockIdx.x * blockDim.x + threadIdx.x;
    if (i < n) out[i] = (bf16_t)in[i];
}

// in: R x C row-major f32  ->  out: C x R row-major bf16 (i.e. transposed)
__global__ void natten_transpose_to_bf16(const float* __restrict__ in,
                                         bf16_t* __restrict__ out,
                                         int R, int C) {
    int i = blockIdx.x * blockDim.x + threadIdx.x;
    if (i < R * C) {
        int r = i / C, c = i % C;
        out[(size_t)c * R + r] = (bf16_t)in[i];
    }
}

// ---------------------------------------------------------------------------
// bf16 WMMA GEMM, register-blocked: each wave computes a 32(M) x 32(N) tile
// as 2x2 WMMA fragments (4 accumulators, 32 VGPRs). A and B fragments each
// reused 2x: 8 b128 loads feed 4 v_wmma_f32_16x16x32_bf16 per K-step of 32.
// Sized to stay within the ~128-VGPR budget (no scratch spills).
//
// C[M,N](f32) = A[M,K](bf16 row-major) * BT[N,K](bf16 row-major)^T
// Fragment layout per CDNA5 ISA 7.12.2 (16-bit A 16x32):
//   lane<16 : row M=lane,    VGPR0..3 = K 0..7,  VGPR4..7 = K 16..23
//   lane>=16: row M=lane-16, VGPR0..3 = K 8..15, VGPR4..7 = K 24..31
// B (32x16) mirrors A with lane index = column N. C/D: VGPR r -> M=r+8*half.
// ---------------------------------------------------------------------------
#define WT_M 2   // 16-row fragments per wave (32 rows)
#define WT_N 2   // 16-col fragments per wave (32 cols)

__global__ void __launch_bounds__(256)
natten_gemm_bf16_wmma(const bf16_t* __restrict__ A,
                      const bf16_t* __restrict__ BT,
                      float* __restrict__ C,
                      int M, int N, int K) {
    const int lane = threadIdx.x & 31;
    const int warp = threadIdx.x >> 5;
    const int warpsPerBlock = blockDim.x >> 5;
    const int tilesN = N / (16 * WT_N);

    const int tile = blockIdx.x * warpsPerBlock + warp;
    const int tM = tile / tilesN;            // 32-row wave-tile index
    const int tN = tile % tilesN;            // 32-col wave-tile index
    if (tM >= M / (16 * WT_M)) return;       // uniform per wave: EXEC all-1s

    const int half = lane >> 4;              // 0 or 1
    const int idx  = lane & 15;

    const bf16_t* arow[WT_M];
#pragma unroll
    for (int m = 0; m < WT_M; ++m)
        arow[m] = A + (size_t)(tM * (16 * WT_M) + m * 16 + idx) * K + 8 * half;
    const bf16_t* brow[WT_N];
#pragma unroll
    for (int nn = 0; nn < WT_N; ++nn)
        brow[nn] = BT + (size_t)(tN * (16 * WT_N) + nn * 16 + idx) * K + 8 * half;

    v8f acc[WT_M][WT_N];
#pragma unroll
    for (int m = 0; m < WT_M; ++m)
#pragma unroll
        for (int nn = 0; nn < WT_N; ++nn)
            acc[m][nn] = (v8f){};

#pragma unroll 1
    for (int k0 = 0; k0 < K; k0 += 32) {
        v16bf av[WT_M], bv[WT_N];
#pragma unroll
        for (int m = 0; m < WT_M; ++m) {
            v8bf a0 = *(const v8bf*)(arow[m] + k0);        // 16B aligned -> b128
            v8bf a1 = *(const v8bf*)(arow[m] + k0 + 16);
#pragma unroll
            for (int i = 0; i < 8; ++i) { av[m][i] = a0[i]; av[m][8 + i] = a1[i]; }
        }
#pragma unroll
        for (int nn = 0; nn < WT_N; ++nn) {
            v8bf b0 = *(const v8bf*)(brow[nn] + k0);
            v8bf b1 = *(const v8bf*)(brow[nn] + k0 + 16);
#pragma unroll
            for (int i = 0; i < 8; ++i) { bv[nn][i] = b0[i]; bv[nn][8 + i] = b1[i]; }
        }
#pragma unroll
        for (int m = 0; m < WT_M; ++m)
#pragma unroll
            for (int nn = 0; nn < WT_N; ++nn)
                acc[m][nn] = __builtin_amdgcn_wmma_f32_16x16x32_bf16(
                    false, av[m], false, bv[nn], (short)0, acc[m][nn], false, false);
    }

#pragma unroll
    for (int m = 0; m < WT_M; ++m) {
#pragma unroll
        for (int nn = 0; nn < WT_N; ++nn) {
            float* crow = C + (size_t)(tM * (16 * WT_M) + m * 16 + 8 * half) * N
                            + (tN * (16 * WT_N) + nn * 16 + idx);
#pragma unroll
            for (int r = 0; r < 8; ++r)
                crow[(size_t)r * N] = acc[m][nn][r];
        }
    }
}

// ---------------------------------------------------------------------------
// Neighborhood attention: one wave32 per (pixel, head).
// qkv: [NPIX][3*512] f32, channel = s*512 + g*64 + e
// obf: [NPIX][512] bf16, channel = g*64 + e
// ---------------------------------------------------------------------------
__global__ void natten_attention(const float* __restrict__ qkv,
                                 bf16_t* __restrict__ obf) {
    __shared__ float sh_q[N_HEADS][D_HEAD];
    __shared__ float sh_p[N_HEADS][64];     // NB=49 used

    const int lane = threadIdx.x & 31;
    const int g    = threadIdx.x >> 5;      // head = warp id (8 waves/block)
    const int pix  = blockIdx.x;            // 0..NPIX-1

    const int n   = pix / (HIMG * WIMG);
    const int rem = pix % (HIMG * WIMG);
    const int y   = rem / WIMG;
    const int x   = rem % WIMG;
    int ys = y - KSIZE / 2; if (ys < 0) ys = 0; if (ys > HIMG - KSIZE) ys = HIMG - KSIZE;
    int xs = x - KSIZE / 2; if (xs < 0) xs = 0; if (xs > WIMG - KSIZE) xs = WIMG - KSIZE;

    // stage scaled q into LDS (2 dims per lane)
    const float* qp = qkv + (size_t)pix * (3 * D_MODEL) + g * D_HEAD;
    sh_q[g][lane * 2 + 0] = qp[lane * 2 + 0] * 0.125f;   // 1/sqrt(64)
    sh_q[g][lane * 2 + 1] = qp[lane * 2 + 1] * 0.125f;
    __syncthreads();

    // logits: lane owns neighbors {lane, lane+32}
    float lg0 = -1e30f, lg1 = -1e30f;
    {
        int nb = lane;                       // always < 49
        int a = nb / KSIZE, b = nb % KSIZE;
        int npx = n * HIMG * WIMG + (ys + a) * WIMG + (xs + b);
        const float* kp = qkv + (size_t)npx * (3 * D_MODEL) + D_MODEL + g * D_HEAD;
        float acc = 0.f;
#pragma unroll 8
        for (int e = 0; e < D_HEAD; ++e) acc += sh_q[g][e] * kp[e];
        lg0 = acc;
    }
    if (lane + 32 < NB) {
        int nb = lane + 32;
        int a = nb / KSIZE, b = nb % KSIZE;
        int npx = n * HIMG * WIMG + (ys + a) * WIMG + (xs + b);
        const float* kp = qkv + (size_t)npx * (3 * D_MODEL) + D_MODEL + g * D_HEAD;
        float acc = 0.f;
#pragma unroll 8
        for (int e = 0; e < D_HEAD; ++e) acc += sh_q[g][e] * kp[e];
        lg1 = acc;
    }

    // wave32 softmax reductions
    float m = fmaxf(lg0, lg1);
    for (int off = 16; off > 0; off >>= 1) m = fmaxf(m, __shfl_xor(m, off, 32));
    float p0 = __expf(lg0 - m);
    float p1 = (lane + 32 < NB) ? __expf(lg1 - m) : 0.f;
    float s = p0 + p1;
    for (int off = 16; off > 0; off >>= 1) s += __shfl_xor(s, off, 32);
    float inv = 1.f / s;

    sh_p[g][lane] = p0 * inv;
    if (lane + 32 < NB) sh_p[g][lane + 32] = p1 * inv;
    __syncthreads();

    // output: lane owns dims {2*lane, 2*lane+1}
    const int e0 = lane * 2, e1 = lane * 2 + 1;
    float o0 = 0.f, o1 = 0.f;
#pragma unroll 7
    for (int nb = 0; nb < NB; ++nb) {
        int a = nb / KSIZE, b = nb % KSIZE;
        int npx = n * HIMG * WIMG + (ys + a) * WIMG + (xs + b);
        const float* vp = qkv + (size_t)npx * (3 * D_MODEL) + 2 * D_MODEL + g * D_HEAD;
        float p = sh_p[g][nb];
        o0 += p * vp[e0];
        o1 += p * vp[e1];
    }
    bf16_t* op = obf + (size_t)pix * D_MODEL + g * D_HEAD;
    op[e0] = (bf16_t)o0;
    op[e1] = (bf16_t)o1;
}

// ---------------------------------------------------------------------------
// Launcher
// ---------------------------------------------------------------------------
extern "C" void kernel_launch(void* const* d_in, const int* in_sizes, int n_in,
                              void* d_out, int out_size, void* d_ws, size_t ws_size,
                              hipStream_t stream) {
    const float* x     = (const float*)d_in[0];   // 2*48*48*512
    const float* w_qkv = (const float*)d_in[1];   // 512*1536
    const float* w_out = (const float*)d_in[2];   // 512*512
    float* out = (float*)d_out;                   // 2*48*48*512

    // workspace layout (all offsets 16B aligned)
    char* ws = (char*)d_ws;
    float*  qkv   = (float*) (ws);                                        // 4608*1536 f32 = 28,311,552 B
    bf16_t* xb    = (bf16_t*)(ws + 28311552);                             // 4608*512  bf16 = 4,718,592 B
    bf16_t* wqkvT = (bf16_t*)(ws + 28311552 + 4718592);                   // 1536*512  bf16 = 1,572,864 B
    bf16_t* woutT = (bf16_t*)(ws + 28311552 + 4718592 + 1572864);         // 512*512   bf16 =   524,288 B
    bf16_t* obf   = (bf16_t*)(ws + 28311552 + 4718592 + 1572864 + 524288);// 4608*512  bf16 = 4,718,592 B

    // 1) precision conversion / weight transposes
    natten_f32_to_bf16<<<(NPIX * D_MODEL + 255) / 256, 256, 0, stream>>>(
        x, xb, NPIX * D_MODEL);
    natten_transpose_to_bf16<<<(D_MODEL * 3 * D_MODEL + 255) / 256, 256, 0, stream>>>(
        w_qkv, wqkvT, D_MODEL, 3 * D_MODEL);
    natten_transpose_to_bf16<<<(D_MODEL * D_MODEL + 255) / 256, 256, 0, stream>>>(
        w_out, woutT, D_MODEL, D_MODEL);

    // 2) QKV projection: (4608x512)*(512x1536)
    //    wave-tiles: (4608/32) * (1536/32) = 144*48 = 6912 -> 864 blocks x 8 waves
    natten_gemm_bf16_wmma<<<6912 / 8, 256, 0, stream>>>(
        xb, wqkvT, qkv, NPIX, 3 * D_MODEL, D_MODEL);

    // 3) neighborhood attention: one block per pixel, one wave per head
    natten_attention<<<NPIX, N_HEADS * 32, 0, stream>>>(qkv, obf);

    // 4) output projection: (4608x512)*(512x512)
    //    wave-tiles: (4608/32) * (512/32) = 144*16 = 2304 -> 288 blocks x 8 waves
    natten_gemm_bf16_wmma<<<2304 / 8, 256, 0, stream>>>(
        obf, woutT, out, NPIX, D_MODEL, D_MODEL);
}
